// Attention_13091060318551
// MI455X (gfx1250) — matrix operand order repaired and checked by reading, not compiled, not run on hardware
//
#include <hip/hip_runtime.h>

// ---------------- WMMA fragment types (wave32, gfx1250) ----------------
typedef __attribute__((ext_vector_type(16))) __bf16 v16bf;
typedef __attribute__((ext_vector_type(8)))  __bf16 v8bf;
typedef __attribute__((ext_vector_type(8)))  float  v8f;

#define WMMA_BF16(a, b, c) \
  __builtin_amdgcn_wmma_f32_16x16x32_bf16(false, (a), false, (b), (short)0, (c), false, false)

static __device__ inline v16bf cat8(v8bf lo, v8bf hi) {
  return __builtin_shufflevector(lo, hi, 0,1,2,3,4,5,6,7,8,9,10,11,12,13,14,15);
}

// A-fragment (16xK=32, 16-bit): lane holds row M=lane&15.
// lanes 0-15: K = 0..7 , 16..23 ;  lanes 16-31: K = 8..15, 24..31  (ISA 7.12.2)
// `row` must already point at A + M*lda + k0.
static __device__ inline v16bf load_frag_a(const __bf16* row, unsigned lane) {
  const int klo = (lane & 16) ? 8 : 0;
  v8bf lo = *(const v8bf*)(row + klo);
  v8bf hi = *(const v8bf*)(row + 16 + klo);
  return cat8(lo, hi);
}

// B-fragment (K=32 x 16, 16-bit): lane holds col N=lane&15.
// lanes 0-15: K = 0..7 , 16..23 ;  lanes 16-31: K = 8..15, 24..31.
// Source is Bt[N][K] row-major; `row` points at Bt + N*ldb + k0.
static __device__ inline v16bf load_frag_b(const __bf16* row, unsigned lane) {
  const int kb = (lane & 16) ? 8 : 0;
  v8bf lo = *(const v8bf*)(row + kb);
  v8bf hi = *(const v8bf*)(row + 16 + kb);
  return cat8(lo, hi);
}

// 16(M) x 64(N) tile GEMM mainloop:  C += A[M,K] * Bt[N,K]^T , K multiple of 32.
static __device__ inline void gemm_tile_16x64(const __bf16* arow /* A + (m0+laneN)*lda */,
                                              const __bf16* Bt, int ldb, int n0, int K,
                                              unsigned lane, v8f acc[4]) {
  const int laneN = lane & 15;
  for (int k0 = 0; k0 < K; k0 += 32) {
    v16bf a = load_frag_a(arow + k0, lane);
#pragma unroll
    for (int n = 0; n < 4; ++n) {
      const __bf16* brow = Bt + (size_t)(n0 + n * 16 + laneN) * ldb + k0;
      v16bf b = load_frag_b(brow, lane);
      acc[n] = WMMA_BF16(a, b, acc[n]);
    }
  }
}

// ---------------- problem constants ----------------
#define BB   2
#define SS   2048
#define DD   1024
#define HH_  16
#define DH   64
#define MTOT (BB * SS)            // 4096 token rows
// inv_freq[j] = 10000^(-j/32) = exp(j * C_FREQ)
#define C_FREQ (-0.28782313662425572f)   // -ln(10000)/32

// ---------------- prep kernels ----------------
__global__ void f32_to_bf16_k(const float* __restrict__ x, __bf16* __restrict__ y, int n) {
  int i = blockIdx.x * blockDim.x + threadIdx.x;
  if (i < n) y[i] = (__bf16)x[i];
}

// W[K][N] (f32) -> Wt[N][K] (bf16)
__global__ void transpose_to_bf16_k(const float* __restrict__ W, __bf16* __restrict__ Wt,
                                    int K, int N) {
  int n = blockIdx.x * blockDim.x + threadIdx.x;
  int k = blockIdx.y * blockDim.y + threadIdx.y;
  if (n < N && k < K) Wt[(size_t)n * K + k] = (__bf16)W[(size_t)k * N + n];
}

// ---------------- QKV GEMM + RoPE + scatter ----------------
// grid: (3072/64, 4096/16), block 32 (one wave). Each wave: 16 tokens x 64 channels,
// which is exactly one (q|k|v, head) dh-block -> RoPE pairs (d, d+-32) are frags n and n^2.
__global__ __launch_bounds__(32) void qkv_rope_kernel(const __bf16* __restrict__ Xbf,
                                                      const __bf16* __restrict__ Wt,
                                                      __bf16* __restrict__ Q,
                                                      __bf16* __restrict__ Kc,
                                                      __bf16* __restrict__ Vt) {
  const unsigned lane = threadIdx.x;
  const int laneN = lane & 15;
  const int halfU = (lane >> 4) & 1;
  const int tb = blockIdx.y * 16;       // token base
  const int cb = blockIdx.x * 64;       // channel base (multiple of dh)

  v8f acc[4] = {};
  const __bf16* arow = Xbf + (size_t)(tb + laneN) * DD;
  gemm_tile_16x64(arow, Wt, DD, cb, DD, lane, acc);

  const int which = cb >> 10;                 // 0=q 1=k 2=v
  const int h = (cb & 1023) >> 6;

  if (which == 2) {
    // V stored transposed: Vt[b][h][d][s]
#pragma unroll
    for (int r = 0; r < 8; ++r) {
      int t = tb + r + halfU * 8;
      int b = t >> 11, s = t & (SS - 1);
      __bf16* dst = Vt + ((size_t)(b * HH_ + h) * DH) * SS + s;
#pragma unroll
      for (int n = 0; n < 4; ++n) {
        int d = n * 16 + laneN;
        dst[(size_t)d * SS] = (__bf16)acc[n][r];
      }
    }
  } else {
    __bf16* Out = (which == 0) ? Q : Kc;      // [b][h][s][d]
#pragma unroll
    for (int r = 0; r < 8; ++r) {
      int t = tb + r + halfU * 8;
      int b = t >> 11, s = t & (SS - 1);
      __bf16* dst = Out + ((size_t)(b * HH_ + h) * SS + s) * DH;
#pragma unroll
      for (int n = 0; n < 4; ++n) {
        int d = n * 16 + laneN;
        int j = d & 31;
        float ang = (float)s * __expf(C_FREQ * (float)j);
        float sn, cs;
        __sincosf(ang, &sn, &cs);
        float partner = (n < 2) ? -acc[n + 2][r] : acc[n - 2][r];  // rotate_half
        dst[d] = (__bf16)(acc[n][r] * cs + partner * sn);
      }
    }
  }
}

// ---------------- flash attention (causal, online softmax) ----------------
// grid: (S/16, H, B), block 32. One wave: 16 queries x full dh=64.
__global__ __launch_bounds__(32) void flash_attn_kernel(const __bf16* __restrict__ Q,
                                                        const __bf16* __restrict__ K,
                                                        const __bf16* __restrict__ Vt,
                                                        __bf16* __restrict__ Z) {
  __shared__ __align__(16) __bf16 Plds[16 * 32];

  const unsigned lane = threadIdx.x;
  const int laneN = lane & 15;
  const int halfU = (lane >> 4) & 1;
  const int qb = blockIdx.x * 16;
  const int h = blockIdx.y, b = blockIdx.z;

  const __bf16* Qh = Q + ((size_t)(b * HH_ + h) * SS) * DH;
  const __bf16* Kh = K + ((size_t)(b * HH_ + h) * SS) * DH;
  const __bf16* Vh = Vt + ((size_t)(b * HH_ + h) * DH) * SS;

  // Q A-fragments for the two d-chunks (d 0..31, 32..63)
  const __bf16* qrow = Qh + (size_t)(qb + laneN) * DH;
  v16bf aq0 = load_frag_a(qrow, lane);
  v16bf aq1 = load_frag_a(qrow + 32, lane);

  v8f o[4] = {};
  float mrow[8], lrow[8];
#pragma unroll
  for (int r = 0; r < 8; ++r) { mrow[r] = -__builtin_inff(); lrow[r] = 0.f; }

  const int kend = qb + 16;       // causal: keys < qb+16
  const float sc = 0.125f;        // 1/sqrt(64)

  for (int kt = 0; kt < kend; kt += 32) {
    // ---- S = Q * K^T for two 16-key subtiles ----
    v8f s0 = {}, s1 = {};
    {
      const __bf16* kr0 = Kh + (size_t)(kt + laneN) * DH;
      s0 = WMMA_BF16(aq0, load_frag_b(kr0, lane), s0);
      s0 = WMMA_BF16(aq1, load_frag_b(kr0 + 32, lane), s0);
      const __bf16* kr1 = Kh + (size_t)(kt + 16 + laneN) * DH;
      s1 = WMMA_BF16(aq0, load_frag_b(kr1, lane), s1);
      s1 = WMMA_BF16(aq1, load_frag_b(kr1 + 32, lane), s1);
    }

    // ---- mask + online softmax (row stats via 16-lane shuffles) ----
#pragma unroll
    for (int r = 0; r < 8; ++r) {
      int q = qb + r + halfU * 8;
      int k0i = kt + laneN, k1i = kt + 16 + laneN;
      float e0 = (k0i <= q) ? s0[r] * sc : -__builtin_inff();
      float e1 = (k1i <= q) ? s1[r] * sc : -__builtin_inff();
      float mx = fmaxf(e0, e1);
      mx = fmaxf(mx, __shfl_xor(mx, 1));
      mx = fmaxf(mx, __shfl_xor(mx, 2));
      mx = fmaxf(mx, __shfl_xor(mx, 4));
      mx = fmaxf(mx, __shfl_xor(mx, 8));
      float mnew = fmaxf(mrow[r], mx);
      float corr = __expf(mrow[r] - mnew);
      mrow[r] = mnew;
      float p0 = __expf(e0 - mnew);
      float p1 = __expf(e1 - mnew);
      float sum = p0 + p1;
      sum += __shfl_xor(sum, 1);
      sum += __shfl_xor(sum, 2);
      sum += __shfl_xor(sum, 4);
      sum += __shfl_xor(sum, 8);
      lrow[r] = lrow[r] * corr + sum;
#pragma unroll
      for (int n = 0; n < 4; ++n) o[n][r] *= corr;
      int M = r + halfU * 8;
      Plds[M * 32 + laneN]      = (__bf16)p0;
      Plds[M * 32 + 16 + laneN] = (__bf16)p1;
    }
    __syncthreads();   // single-wave WG: cheap; orders LDS RAW

    // ---- relayout P (C-layout -> A-fragment) via LDS ----
    v16bf ap = load_frag_a(Plds + laneN * 32, lane);
    __syncthreads();

    // ---- O += P * V  (V stored transposed -> contiguous B-fragments) ----
#pragma unroll
    for (int n = 0; n < 4; ++n) {
      const __bf16* vr = Vh + (size_t)(n * 16 + laneN) * SS + kt;
      o[n] = WMMA_BF16(ap, load_frag_b(vr, lane), o[n]);
    }
  }

  // ---- normalize and write Z[b][s][h*64+d] (bf16, token-major for out GEMM) ----
  __bf16* Zb = Z + ((size_t)b * SS) * DD + (size_t)h * DH;
#pragma unroll
  for (int r = 0; r < 8; ++r) {
    int q = qb + r + halfU * 8;
    float inv = 1.0f / lrow[r];
    __bf16* dst = Zb + (size_t)q * DD;
#pragma unroll
    for (int n = 0; n < 4; ++n)
      dst[n * 16 + laneN] = (__bf16)(o[n][r] * inv);
  }
}

// ---------------- output projection GEMM: out = Z @ W_o (f32 out) ----------------
__global__ __launch_bounds__(32) void out_gemm_kernel(const __bf16* __restrict__ Zbf,
                                                      const __bf16* __restrict__ Wot,
                                                      float* __restrict__ out) {
  const unsigned lane = threadIdx.x;
  const int laneN = lane & 15;
  const int halfU = (lane >> 4) & 1;
  const int tb = blockIdx.y * 16;
  const int nb = blockIdx.x * 64;

  v8f acc[4] = {};
  const __bf16* arow = Zbf + (size_t)(tb + laneN) * DD;
  gemm_tile_16x64(arow, Wot, DD, nb, DD, lane, acc);

#pragma unroll
  for (int r = 0; r < 8; ++r) {
    float* dst = out + (size_t)(tb + r + halfU * 8) * DD + nb;
#pragma unroll
    for (int n = 0; n < 4; ++n) dst[n * 16 + laneN] = acc[n][r];
  }
}

// ---------------- launch ----------------
extern "C" void kernel_launch(void* const* d_in, const int* in_sizes, int n_in,
                              void* d_out, int out_size, void* d_ws, size_t ws_size,
                              hipStream_t stream) {
  (void)in_sizes; (void)n_in; (void)out_size; (void)ws_size;
  const float* x    = (const float*)d_in[0];   // [2,2048,1024]
  const float* Wqkv = (const float*)d_in[1];   // [1024,3072]
  const float* Wo   = (const float*)d_in[2];   // [1024,1024]
  float* out = (float*)d_out;                  // [2,2048,1024]

  char* ws = (char*)d_ws;
  const size_t MB = 1u << 20;
  __bf16* Xbf   = (__bf16*)(ws + 0 * MB);   // 8 MB  [4096,1024]
  __bf16* Wqkvt = (__bf16*)(ws + 8 * MB);   // 6 MB  [3072,1024]
  __bf16* Wot   = (__bf16*)(ws + 14 * MB);  // 2 MB  [1024,1024]
  __bf16* Qb    = (__bf16*)(ws + 16 * MB);  // 8 MB  [B,H,S,dh]
  __bf16* Kb    = (__bf16*)(ws + 24 * MB);  // 8 MB  [B,H,S,dh]
  __bf16* Vt    = (__bf16*)(ws + 32 * MB);  // 8 MB  [B,H,dh,S]
  __bf16* Zb    = (__bf16*)(ws + 40 * MB);  // 8 MB  [B,S,D]

  const int nx = MTOT * DD;
  f32_to_bf16_k<<<(nx + 255) / 256, 256, 0, stream>>>(x, Xbf, nx);
  transpose_to_bf16_k<<<dim3(3 * DD / 16, DD / 16), dim3(16, 16), 0, stream>>>(Wqkv, Wqkvt, DD, 3 * DD);
  transpose_to_bf16_k<<<dim3(DD / 16, DD / 16), dim3(16, 16), 0, stream>>>(Wo, Wot, DD, DD);

  qkv_rope_kernel<<<dim3(3 * DD / 64, MTOT / 16), 32, 0, stream>>>(Xbf, Wqkvt, Qb, Kb, Vt);
  flash_attn_kernel<<<dim3(SS / 16, HH_, BB), 32, 0, stream>>>(Qb, Kb, Vt, Zb);
  out_gemm_kernel<<<dim3(DD / 64, MTOT / 16), 32, 0, stream>>>(Zb, Wot, out);
}
